// R_TuckER_30021821399332
// MI455X (gfx1250) — matrix operand order repaired and checked by reading, not compiled, not run on hardware
//
#include <hip/hip_runtime.h>

// ---------------------------------------------------------------------------
// TuckER forward for MI455X (gfx1250, wave32, WMMA bf16 16x16x32, f32 accum)
//   preds = sigmoid( ((x ⊗ rel) @ core_flat) @ S^T )
// B=512, DE=DR=256, N_ENT=100000, N_REL=500
//
// Roofline: ~43.5 GFLOP, HBM floor ~370 MB (core 64MB + S 102MB + out 205MB)
// ≈ 16 us @ 23.3 TB/s.  bf16 WMMA makes this memory-bound; core and S_weight
// are L2-resident (192 MB), so split-K / tile re-reads are L2 traffic only.
//
// Fast path (gated on ws_size): core transposed+converted once to bf16 so the
// Tucker GEMM inner loop is LDS/barrier-free (contiguous 32B B-fragments,
// register-built A fragments); S_weight pre-converted to bf16 likewise.
// ---------------------------------------------------------------------------

typedef __attribute__((ext_vector_type(16))) __bf16 v16bf;
typedef __attribute__((ext_vector_type(8)))  float  v8f;

union Frag16 {                    // one 16x16x32 bf16 WMMA operand per lane
  v16bf v;
  struct { uint4 lo; uint4 hi; } q;   // two 16-byte chunks
};

static __device__ __forceinline__ v8f wmma_bf16(const v16bf& a, const v16bf& b, const v8f& c) {
  return __builtin_amdgcn_wmma_f32_16x16x32_bf16(
      /*neg_a=*/false, a, /*neg_b=*/false, b,
      /*c_mod=*/(short)0, c, /*reuse_a=*/false, /*reuse_b=*/false);
}

#define N_ENT 100000
#define KCHUNK 1024

// ---------------------------------------------------------------------------
// Kernel 1: embedding gather + BatchNorm1d (batch stats, biased variance) +
// convert to bf16.  One block per feature column; 512-row reduction.
// ---------------------------------------------------------------------------
__global__ __launch_bounds__(256)
void bn_gather_kernel(const int* __restrict__ idx, const float* __restrict__ W,
                      const float* __restrict__ gamma, const float* __restrict__ beta,
                      __bf16* __restrict__ out_bf) {
  const int f = blockIdx.x;          // feature 0..255
  const int t = threadIdx.x;         // 0..255, each handles rows t and t+256
  __shared__ float s_sum[256];
  __shared__ float s_sq[256];
  __shared__ float s_scale, s_shift;

  const float v0 = W[(size_t)idx[t]       * 256 + f];
  const float v1 = W[(size_t)idx[t + 256] * 256 + f];
  s_sum[t] = v0 + v1;
  s_sq[t]  = v0 * v0 + v1 * v1;
  __syncthreads();
  for (int off = 128; off > 0; off >>= 1) {
    if (t < off) { s_sum[t] += s_sum[t + off]; s_sq[t] += s_sq[t + off]; }
    __syncthreads();
  }
  if (t == 0) {
    const float mean = s_sum[0] * (1.0f / 512.0f);
    const float var  = s_sq[0] * (1.0f / 512.0f) - mean * mean;   // biased
    const float scl  = gamma[f] * rsqrtf(var + 1e-5f);
    s_scale = scl;
    s_shift = beta[f] - mean * scl;
  }
  __syncthreads();
  out_bf[(size_t)t * 256 + f]         = (__bf16)(v0 * s_scale + s_shift);
  out_bf[(size_t)(t + 256) * 256 + f] = (__bf16)(v1 * s_scale + s_shift);
}

// ---------------------------------------------------------------------------
// Kernel 2: zero the split-K f32 accumulator buffer.
// ---------------------------------------------------------------------------
__global__ void zero_f32_kernel(float* __restrict__ p, int n) {
  const int i = blockIdx.x * blockDim.x + threadIdx.x;
  if (i < n) p[i] = 0.0f;
}

// ---------------------------------------------------------------------------
// Kernel 3a (fast path): one-time core transpose + bf16 convert.
//   core   f32 [65536 K][256 n]  ->  coreT bf16 [256 n][65536 K]
// LDS-tiled 32x32 transpose; writes 8B packed bf16x4 per thread per tile.
// ---------------------------------------------------------------------------
__global__ __launch_bounds__(256)
void core_transpose_kernel(const float* __restrict__ core, __bf16* __restrict__ coreT) {
  __shared__ float tile[32][33];                // +1 pad: kill bank conflicts
  const int K0 = blockIdx.x * 32;               // 2048 tiles along K
  const int n0 = blockIdx.y * 32;               // 8 tiles along n
  const int t  = threadIdx.x;
  {
    const int r = t >> 3;                       // 0..31 (K row)
    const int c = (t & 7) * 4;                  // 0..28 (n col)
    const float4 v = *(const float4*)(core + (size_t)(K0 + r) * 256 + n0 + c);
    tile[r][c] = v.x; tile[r][c + 1] = v.y; tile[r][c + 2] = v.z; tile[r][c + 3] = v.w;
  }
  __syncthreads();
  {
    const int rn = t >> 3;                      // 0..31 (n row of coreT)
    const int ck = (t & 7) * 4;                 // 0..28 (K col)
    union { __bf16 h[4]; unsigned long long u; } p;
    p.h[0] = (__bf16)tile[ck][rn];
    p.h[1] = (__bf16)tile[ck + 1][rn];
    p.h[2] = (__bf16)tile[ck + 2][rn];
    p.h[3] = (__bf16)tile[ck + 3][rn];
    *(unsigned long long*)(coreT + (size_t)(n0 + rn) * 65536 + K0 + ck) = p.u;
  }
}

// ---------------------------------------------------------------------------
// Kernel 3b (fast path): fused Tucker GEMM, LDS-free / barrier-free.
//   out2[512,256] += Z[512,65536] @ core,  Z[b,i*256+j] = x[b,i]*rel[b,j]
// A fragments built in registers (rel chunk * hoisted x scalar); B fragments
// are contiguous 32-byte loads from coreT (L2-resident).  Split-K atomics.
// ---------------------------------------------------------------------------
__global__ __launch_bounds__(128)
void tucker_gemm_fast_kernel(const __bf16* __restrict__ xbf, const __bf16* __restrict__ relbf,
                             const __bf16* __restrict__ coreT, float* __restrict__ out2) {
  const int tid  = threadIdx.x;
  const int lane = tid & 31;
  const int wave = tid >> 5;
  const int nb = blockIdx.x * 64;
  const int mb = blockIdx.y * 64;
  const int kbase = blockIdx.z * KCHUNK;
  const int wm = (wave >> 1) * 32;
  const int wn = (wave & 1) * 32;
  const int mrow = lane & 15;
  const int ksel = lane >> 4;

  const int m0 = mb + wm + mrow, m1 = m0 + 16;  // A rows for this lane
  const int n0 = nb + wn + mrow, n1 = n0 + 16;  // B cols (coreT rows)

  const __bf16* relp0 = relbf + (size_t)m0 * 256;
  const __bf16* relp1 = relbf + (size_t)m1 * 256;
  const __bf16* bp0 = coreT + (size_t)n0 * 65536 + kbase + 16 * ksel;
  const __bf16* bp1 = coreT + (size_t)n1 * 65536 + kbase + 16 * ksel;

  v8f c00 = {}, c01 = {}, c10 = {}, c11 = {};

  for (int ii = 0; ii < KCHUNK / 256; ++ii) {   // i (x index) hoisted
    const int i = (kbase >> 8) + ii;
    const float x0 = (float)xbf[(size_t)m0 * 256 + i];
    const float x1 = (float)xbf[(size_t)m1 * 256 + i];
#pragma unroll
    for (int jj = 0; jj < 8; ++jj) {            // 8 x 32-K steps per i
      const int j0 = jj * 32;
      Frag16 a0, a1, b0, b1;
      const uint4* r0 = (const uint4*)(relp0 + j0);
      const uint4* r1 = (const uint4*)(relp1 + j0);
      a0.q.lo = r0[2 * ksel]; a0.q.hi = r0[2 * ksel + 1];   // K 16s..16s+15
      a1.q.lo = r1[2 * ksel]; a1.q.hi = r1[2 * ksel + 1];
#pragma unroll
      for (int u = 0; u < 16; ++u) {            // Z = x * rel (per-lane, in regs)
        a0.v[u] = (__bf16)((float)a0.v[u] * x0);
        a1.v[u] = (__bf16)((float)a1.v[u] * x1);
      }
      const uint4* bq0 = (const uint4*)(bp0 + ii * 256 + j0);
      const uint4* bq1 = (const uint4*)(bp1 + ii * 256 + j0);
      b0.q.lo = bq0[0]; b0.q.hi = bq0[1];       // 16 contiguous K halves
      b1.q.lo = bq1[0]; b1.q.hi = bq1[1];

      c00 = wmma_bf16(a0.v, b0.v, c00);
      c01 = wmma_bf16(a0.v, b1.v, c01);
      c10 = wmma_bf16(a1.v, b0.v, c10);
      c11 = wmma_bf16(a1.v, b1.v, c11);
    }
  }

  const int n_l = lane & 15;
  const int mhi = (lane >> 4) * 8;
#pragma unroll
  for (int r = 0; r < 8; ++r) {
    const int m = mhi + r;
    atomicAdd(&out2[(size_t)(mb + wm + m)      * 256 + nb + wn + n_l],      c00[r]);
    atomicAdd(&out2[(size_t)(mb + wm + m)      * 256 + nb + wn + 16 + n_l], c01[r]);
    atomicAdd(&out2[(size_t)(mb + wm + 16 + m) * 256 + nb + wn + n_l],      c10[r]);
    atomicAdd(&out2[(size_t)(mb + wm + 16 + m) * 256 + nb + wn + 16 + n_l], c11[r]);
  }
}

// ---------------------------------------------------------------------------
// Kernel 3c (fallback): Tucker GEMM with per-tile LDS transpose of core.
// ---------------------------------------------------------------------------
__global__ __launch_bounds__(128)
void tucker_gemm_kernel(const __bf16* __restrict__ xbf, const __bf16* __restrict__ relbf,
                        const float* __restrict__ core, float* __restrict__ out2) {
  __shared__ __align__(32) __bf16 As[64][32];   // Z tile, row-major [m][k]
  __shared__ __align__(32) __bf16 Bs[64][32];   // core tile, col-major [n][k]

  const int tid  = threadIdx.x;
  const int lane = tid & 31;
  const int wave = tid >> 5;
  const int nb = blockIdx.x * 64;
  const int mb = blockIdx.y * 64;
  const int kbase = blockIdx.z * KCHUNK;
  const int wm = (wave >> 1) * 32;
  const int wn = (wave & 1) * 32;
  const int mrow = lane & 15;
  const int ksel = lane >> 4;

  v8f c00 = {}, c01 = {}, c10 = {}, c11 = {};

  for (int kt = 0; kt < KCHUNK; kt += 32) {
    const int K0 = kbase + kt;
    const int i  = K0 >> 8;
    const int j0 = K0 & 255;
    __syncthreads();
    {
      const int m  = tid >> 1;
      const int kh = (tid & 1) * 16;
      const int b  = mb + m;
      const float xv = (float)xbf[(size_t)b * 256 + i];
      const __bf16* rp = relbf + (size_t)b * 256 + j0 + kh;
#pragma unroll
      for (int u = 0; u < 16; ++u)
        As[m][kh + u] = (__bf16)(xv * (float)rp[u]);
    }
    {
      const int k = tid >> 2;
      const int n = (tid & 3) * 16;
      const float* gp = core + (size_t)(K0 + k) * 256 + nb + n;
      __builtin_prefetch(gp + 32 * 256, 0, 0);
#pragma unroll
      for (int u = 0; u < 16; ++u)
        Bs[n + u][k] = (__bf16)gp[u];
    }
    __syncthreads();

    Frag16 a0, a1, b0, b1;
    const uint4* ar0 = (const uint4*)&As[wm + mrow][0];
    const uint4* ar1 = (const uint4*)&As[wm + 16 + mrow][0];
    a0.q.lo = ar0[2 * ksel]; a0.q.hi = ar0[2 * ksel + 1];
    a1.q.lo = ar1[2 * ksel]; a1.q.hi = ar1[2 * ksel + 1];
    const uint4* br0 = (const uint4*)&Bs[wn + mrow][0];
    const uint4* br1 = (const uint4*)&Bs[wn + 16 + mrow][0];
    b0.q.lo = br0[2 * ksel]; b0.q.hi = br0[2 * ksel + 1];
    b1.q.lo = br1[2 * ksel]; b1.q.hi = br1[2 * ksel + 1];

    c00 = wmma_bf16(a0.v, b0.v, c00);
    c01 = wmma_bf16(a0.v, b1.v, c01);
    c10 = wmma_bf16(a1.v, b0.v, c10);
    c11 = wmma_bf16(a1.v, b1.v, c11);
  }

  const int n_l = lane & 15;
  const int mhi = (lane >> 4) * 8;
#pragma unroll
  for (int r = 0; r < 8; ++r) {
    const int m = mhi + r;
    atomicAdd(&out2[(size_t)(mb + wm + m)      * 256 + nb + wn + n_l],      c00[r]);
    atomicAdd(&out2[(size_t)(mb + wm + m)      * 256 + nb + wn + 16 + n_l], c01[r]);
    atomicAdd(&out2[(size_t)(mb + wm + 16 + m) * 256 + nb + wn + n_l],      c10[r]);
    atomicAdd(&out2[(size_t)(mb + wm + 16 + m) * 256 + nb + wn + 16 + n_l], c11[r]);
  }
}

// ---------------------------------------------------------------------------
// Kernel 4: f32 -> bf16 converts (scalar and x4-vectorized variants).
// ---------------------------------------------------------------------------
__global__ void cvt_bf16_kernel(const float* __restrict__ in, __bf16* __restrict__ out, int n) {
  const int i = blockIdx.x * blockDim.x + threadIdx.x;
  if (i < n) out[i] = (__bf16)in[i];
}

__global__ void cvt_bf16_x4_kernel(const float* __restrict__ in, __bf16* __restrict__ out, int n4) {
  const int i = blockIdx.x * blockDim.x + threadIdx.x;
  if (i < n4) {
    const float4 v = ((const float4*)in)[i];
    union { __bf16 h[4]; unsigned long long u; } p;
    p.h[0] = (__bf16)v.x; p.h[1] = (__bf16)v.y; p.h[2] = (__bf16)v.z; p.h[3] = (__bf16)v.w;
    ((unsigned long long*)out)[i] = p.u;
  }
}

// ---------------------------------------------------------------------------
// Kernel 5a (fast path): logits GEMM + sigmoid with pre-converted S_bf.
// B fragment = 32 contiguous bytes of S_bf row e.  Edge tiles clamp/guard.
// ---------------------------------------------------------------------------
__global__ __launch_bounds__(128)
void logits_gemm_fast_kernel(const __bf16* __restrict__ t2bf, const __bf16* __restrict__ Sbf,
                             float* __restrict__ out) {
  const int tid  = threadIdx.x;
  const int lane = tid & 31;
  const int wave = tid >> 5;
  const int nb = blockIdx.x * 64;
  const int mb = blockIdx.y * 64;
  const int wm = (wave >> 1) * 32;
  const int wn = (wave & 1) * 32;
  const int mrow = lane & 15;
  const int ksel = lane >> 4;

  const int m0g = mb + wm + mrow;
  const int m1g = m0g + 16;
  const int e0 = nb + wn + mrow;
  const int e1 = e0 + 16;
  const int e0c = (e0 < N_ENT) ? e0 : (N_ENT - 1);
  const int e1c = (e1 < N_ENT) ? e1 : (N_ENT - 1);

  v8f c00 = {}, c01 = {}, c10 = {}, c11 = {};

#pragma unroll
  for (int K0 = 0; K0 < 256; K0 += 32) {
    Frag16 a0, a1, b0, b1;
    const uint4* ar0 = (const uint4*)(t2bf + (size_t)m0g * 256 + K0);
    const uint4* ar1 = (const uint4*)(t2bf + (size_t)m1g * 256 + K0);
    a0.q.lo = ar0[2 * ksel]; a0.q.hi = ar0[2 * ksel + 1];
    a1.q.lo = ar1[2 * ksel]; a1.q.hi = ar1[2 * ksel + 1];
    const uint4* sp0 = (const uint4*)(Sbf + (size_t)e0c * 256 + K0 + ksel * 16);
    const uint4* sp1 = (const uint4*)(Sbf + (size_t)e1c * 256 + K0 + ksel * 16);
    b0.q.lo = sp0[0]; b0.q.hi = sp0[1];
    b1.q.lo = sp1[0]; b1.q.hi = sp1[1];

    c00 = wmma_bf16(a0.v, b0.v, c00);
    c01 = wmma_bf16(a0.v, b1.v, c01);
    c10 = wmma_bf16(a1.v, b0.v, c10);
    c11 = wmma_bf16(a1.v, b1.v, c11);
  }

  const int n_l = lane & 15;
  const int mhi = (lane >> 4) * 8;
  const int ea  = nb + wn + n_l;
  const int ebb = ea + 16;
#pragma unroll
  for (int r = 0; r < 8; ++r) {
    const int ma  = mb + wm + mhi + r;
    const int mbx = ma + 16;
    if (ea < N_ENT) {
      out[(size_t)ma  * N_ENT + ea] = 1.0f / (1.0f + __expf(-c00[r]));
      out[(size_t)mbx * N_ENT + ea] = 1.0f / (1.0f + __expf(-c10[r]));
    }
    if (ebb < N_ENT) {
      out[(size_t)ma  * N_ENT + ebb] = 1.0f / (1.0f + __expf(-c01[r]));
      out[(size_t)mbx * N_ENT + ebb] = 1.0f / (1.0f + __expf(-c11[r]));
    }
  }
}

// ---------------------------------------------------------------------------
// Kernel 5b (fallback): logits GEMM + sigmoid, inline f32->bf16 cvt of S.
// ---------------------------------------------------------------------------
__global__ __launch_bounds__(128)
void logits_gemm_kernel(const __bf16* __restrict__ t2bf, const float* __restrict__ S,
                        float* __restrict__ out) {
  const int tid  = threadIdx.x;
  const int lane = tid & 31;
  const int wave = tid >> 5;
  const int nb = blockIdx.x * 64;
  const int mb = blockIdx.y * 64;
  const int wm = (wave >> 1) * 32;
  const int wn = (wave & 1) * 32;
  const int mrow = lane & 15;
  const int ksel = lane >> 4;

  const int m0g = mb + wm + mrow;
  const int m1g = m0g + 16;
  const int e0 = nb + wn + mrow;
  const int e1 = e0 + 16;
  const int e0c = (e0 < N_ENT) ? e0 : (N_ENT - 1);
  const int e1c = (e1 < N_ENT) ? e1 : (N_ENT - 1);

  v8f c00 = {}, c01 = {}, c10 = {}, c11 = {};

#pragma unroll
  for (int K0 = 0; K0 < 256; K0 += 32) {
    Frag16 a0, a1, b0, b1;
    const uint4* ar0 = (const uint4*)(t2bf + (size_t)m0g * 256 + K0);
    const uint4* ar1 = (const uint4*)(t2bf + (size_t)m1g * 256 + K0);
    a0.q.lo = ar0[2 * ksel]; a0.q.hi = ar0[2 * ksel + 1];
    a1.q.lo = ar1[2 * ksel]; a1.q.hi = ar1[2 * ksel + 1];
    const float* sp0 = S + (size_t)e0c * 256 + K0 + ksel * 16;
    const float* sp1 = S + (size_t)e1c * 256 + K0 + ksel * 16;
#pragma unroll
    for (int u = 0; u < 16; ++u) {
      b0.v[u] = (__bf16)sp0[u];
      b1.v[u] = (__bf16)sp1[u];
    }
    c00 = wmma_bf16(a0.v, b0.v, c00);
    c01 = wmma_bf16(a0.v, b1.v, c01);
    c10 = wmma_bf16(a1.v, b0.v, c10);
    c11 = wmma_bf16(a1.v, b1.v, c11);
  }

  const int n_l = lane & 15;
  const int mhi = (lane >> 4) * 8;
  const int ea  = nb + wn + n_l;
  const int ebb = ea + 16;
#pragma unroll
  for (int r = 0; r < 8; ++r) {
    const int ma  = mb + wm + mhi + r;
    const int mbx = ma + 16;
    if (ea < N_ENT) {
      out[(size_t)ma  * N_ENT + ea] = 1.0f / (1.0f + __expf(-c00[r]));
      out[(size_t)mbx * N_ENT + ea] = 1.0f / (1.0f + __expf(-c10[r]));
    }
    if (ebb < N_ENT) {
      out[(size_t)ma  * N_ENT + ebb] = 1.0f / (1.0f + __expf(-c01[r]));
      out[(size_t)mbx * N_ENT + ebb] = 1.0f / (1.0f + __expf(-c11[r]));
    }
  }
}

// ---------------------------------------------------------------------------
// Host launcher.  ws layout:
//   [0]        x_bf    512*256 bf16   (256 KB)
//   [256K]     rel_bf  512*256 bf16   (256 KB)
//   [512K]     out2    512*256 f32    (512 KB)
//   [1M]       t2_bf   512*256 bf16   (256 KB)
//   [1.25M]    coreT   256*65536 bf16 (32 MB)    -- fast path only
//   [~33.25M]  S_bf    100000*256 bf16 (48.8 MB) -- fast path only
// Gating on ws_size is deterministic (fixed per process).
// ---------------------------------------------------------------------------
extern "C" void kernel_launch(void* const* d_in, const int* in_sizes, int n_in,
                              void* d_out, int out_size, void* d_ws, size_t ws_size,
                              hipStream_t stream) {
  (void)in_sizes; (void)n_in; (void)out_size;
  const int*   subject_idx  = (const int*)d_in[0];
  const int*   relation_idx = (const int*)d_in[1];
  const float* S_weight     = (const float*)d_in[2];
  const float* R_weight     = (const float*)d_in[3];
  const float* core         = (const float*)d_in[4];
  const float* bn0_gamma    = (const float*)d_in[5];
  const float* bn0_beta     = (const float*)d_in[6];
  const float* bn1_gamma    = (const float*)d_in[7];
  const float* bn1_beta     = (const float*)d_in[8];
  float* out = (float*)d_out;

  char* ws = (char*)d_ws;
  const size_t OFF_CORET  = 1310720;                          // 1.25 MB
  const size_t CORET_B    = (size_t)256 * 65536 * 2;          // 33,554,432
  const size_t OFF_SBF    = OFF_CORET + CORET_B;              // 34,865,152
  const size_t SBF_B      = (size_t)N_ENT * 256 * 2;          // 51,200,000
  const bool fast_core = ws_size >= OFF_SBF;
  const bool fast_S    = ws_size >= OFF_SBF + SBF_B;

  __bf16* x_bf   = (__bf16*)(ws);
  __bf16* rel_bf = (__bf16*)(ws + 262144);
  float*  out2   = (float*) (ws + 524288);
  __bf16* t2_bf  = (__bf16*)(ws + 1048576);
  __bf16* coreT  = (__bf16*)(ws + OFF_CORET);
  __bf16* S_bf   = (__bf16*)(ws + OFF_SBF);

  // 1) gather + batchnorm (bn0 on subjects, bn1 on relations) -> bf16
  bn_gather_kernel<<<256, 256, 0, stream>>>(subject_idx, S_weight, bn0_gamma, bn0_beta, x_bf);
  bn_gather_kernel<<<256, 256, 0, stream>>>(relation_idx, R_weight, bn1_gamma, bn1_beta, rel_bf);

  // 2) zero split-K accumulator
  zero_f32_kernel<<<512, 256, 0, stream>>>(out2, 512 * 256);

  // 3) fused (x ⊗ rel) @ core_flat  [512 x 65536] @ [65536 x 256]
  if (fast_core) {
    core_transpose_kernel<<<dim3(2048, 8), 256, 0, stream>>>(core, coreT);
    tucker_gemm_fast_kernel<<<dim3(4, 8, 64), 128, 0, stream>>>(x_bf, rel_bf, coreT, out2);
  } else {
    tucker_gemm_kernel<<<dim3(4, 8, 64), 128, 0, stream>>>(x_bf, rel_bf, core, out2);
  }

  // 4) t2 -> bf16
  cvt_bf16_kernel<<<512, 256, 0, stream>>>(out2, t2_bf, 512 * 256);

  // 5) logits = t2 @ S^T, sigmoid   [512 x 256] @ [256 x 100000]
  if (fast_S) {
    cvt_bf16_x4_kernel<<<25000, 256, 0, stream>>>(S_weight, S_bf, N_ENT * 256 / 4);
    logits_gemm_fast_kernel<<<dim3((N_ENT + 63) / 64, 8), 128, 0, stream>>>(t2_bf, S_bf, out);
  } else {
    logits_gemm_kernel<<<dim3((N_ENT + 63) / 64, 8), 128, 0, stream>>>(t2_bf, S_weight, out);
  }
}
